// Encoder_Block_28570122453038
// MI455X (gfx1250) — compile-verified, matrix operand order checked by reading, not run on hardware
//
#include <hip/hip_runtime.h>
#include <hip/hip_bf16.h>

typedef __bf16 bf16;
typedef __attribute__((ext_vector_type(16))) __bf16 v16bf;
typedef __attribute__((ext_vector_type(2)))  __bf16 bf16x2;
typedef __attribute__((ext_vector_type(8)))  float  v8f;

#define DIM   256
#define NSEQ  128
#define BATCH 8
#define HID   1024

// ---------------------------------------------------------------- WMMA utils

__device__ inline v8f wmma_bf16(v16bf a, v16bf b, v8f c) {
  // D = A(16x32) * B(32x16) + C, f32 accumulate
  return __builtin_amdgcn_wmma_f32_16x16x32_bf16(false, a, false, b, (short)0, c,
                                                 false, false);
}

// A-fragment from LDS, bf16 row-major, row stride DIM.
// 16-bit A 16x32 layout (ISA 7.12.2): lane L holds row M=L%16;
// lanes 0-15: K = {0..7, 16..23}; lanes 16-31: K = {8..15, 24..31};
// VGPR p holds K pair ((p>=4)?16:0) + khalf + (p%4)*2 + {0,1}.
__device__ inline v16bf lds_afrag(const bf16* sA, int row0, int k0) {
  const int lane = threadIdx.x & 31;
  const bf16* base = sA + (row0 + (lane & 15)) * DIM + k0 + ((lane & 16) >> 1);
  v16bf a;
#pragma unroll
  for (int p = 0; p < 8; ++p) {
    bf16x2 two = *(const bf16x2*)(base + ((p & 4) << 2) + ((p & 3) << 1));
    a[2 * p]     = two.x;
    a[2 * p + 1] = two.y;
  }
  return a;
}

// B-fragment: weights pre-packed so each fragment is 32 lanes x 16 bf16
// contiguous -> one 32B/lane coalesced load (2x global_load_b128).
__device__ inline v16bf gbl_bfrag(const bf16* __restrict__ pack, int frag) {
  const int lane = threadIdx.x & 31;
  return *(const v16bf*)(pack + ((size_t)frag << 9) + ((size_t)lane << 4));
}

// Pipelined 128xDIM-row-tile GEMM: this wave computes row-tile `wave`,
// all 16 col-tiles. A fragments preloaded to registers. B fragments flow
// through a 3-deep rotating buffer (w0,w1,w2 all live across each WMMA, so
// the allocator cannot collapse the pipeline): each fragment is issued 3
// WMMA-slots before consumption -> s_wait_loadcnt > 0 at each WMMA and L2
// latency overlaps with matrix math. frag(nt,kt) = fragbase + nt*fragstride + kt.
template <typename FInit, typename FStore>
__device__ inline void gemm_16tiles(const bf16* __restrict__ sAsrc,
                                    const bf16* __restrict__ pack,
                                    int fragbase, int fragstride,
                                    FInit finit, FStore fstore) {
  const int wave = threadIdx.x >> 5;
  v16bf a[8];
#pragma unroll
  for (int kt = 0; kt < 8; ++kt) a[kt] = lds_afrag(sAsrc, wave * 16, kt * 32);

  auto fragof = [&](int i) {
    i = (i > 127) ? 127 : i;                       // clamp tail lookahead
    return fragbase + (i >> 3) * fragstride + (i & 7);
  };
  v16bf w0 = gbl_bfrag(pack, fragof(0));
  v16bf w1 = gbl_bfrag(pack, fragof(1));
  v16bf w2 = gbl_bfrag(pack, fragof(2));
  for (int nt = 0; nt < 16; ++nt) {
    v8f acc = finit(nt);
#pragma unroll
    for (int kt = 0; kt < 8; ++kt) {
      v16bf wn = gbl_bfrag(pack, fragof(nt * 8 + kt + 3)); // issue 3 ahead
      acc = wmma_bf16(a[kt], w0, acc);                     // consume oldest
      w0 = w1; w1 = w2; w2 = wn;
    }
    fstore(nt, acc);
  }
}

// ------------------------------------------------- weight pre-pack (f32->bf16)
// pack[frag][lane][t], frag = nt*(K/32)+kt; B lane L holds col N=L%16 with the
// K pattern mirroring the A layout above.
__global__ void pack_kernel(const float* __restrict__ W, bf16* __restrict__ dst,
                            int K, int N) {
  int tid  = blockIdx.x * 256 + threadIdx.x;
  int nk   = K >> 5;
  int frag = tid >> 9;
  int r    = tid & 511;
  int lane = r >> 4;
  int t    = r & 15;
  int nt   = frag / nk;
  int kt   = frag - nt * nk;
  int n    = (nt << 4) + (lane & 15);
  int p    = t >> 1;
  int k    = (kt << 5) + ((p & 4) << 2) + ((lane & 16) >> 1) + ((p & 3) << 1) + (t & 1);
  dst[tid] = (bf16)W[(size_t)k * N + n];
}

// ------------------------------------------------------------- x1 = LN(x)
__global__ void lnx_kernel(const float* __restrict__ x, const float* __restrict__ g,
                           const float* __restrict__ bb, float* __restrict__ x1) {
  int row  = blockIdx.x * 8 + (threadIdx.x >> 5);
  int lane = threadIdx.x & 31;
  const float* xr = x + (size_t)row * DIM;
  float v[8], s = 0.f, s2 = 0.f;
#pragma unroll
  for (int cc = 0; cc < 8; ++cc) {
    float t = xr[lane + cc * 32];
    v[cc] = t; s += t; s2 += t * t;
  }
#pragma unroll
  for (int m = 16; m >= 1; m >>= 1) {
    s  += __shfl_xor(s, m, 32);
    s2 += __shfl_xor(s2, m, 32);
  }
  float mean = s * (1.f / DIM);
  float var  = s2 * (1.f / DIM) - mean * mean;
  float rs   = rsqrtf(var + 1e-5f);
  float* o = x1 + (size_t)row * DIM;
#pragma unroll
  for (int cc = 0; cc < 8; ++cc) {
    int c = lane + cc * 32;
    o[c] = (v[cc] - mean) * rs * g[c] + bb[c];
  }
}

// --------------------------------------------------- q/k/v = x1 @ W + b (WMMA)
__global__ __launch_bounds__(256)
void qkv_kernel(const float* __restrict__ x1,
                const bf16* __restrict__ pq, const bf16* __restrict__ pk,
                const bf16* __restrict__ pv,
                const float* __restrict__ bq, const float* __restrict__ bk,
                const float* __restrict__ bv,
                float* __restrict__ qo, float* __restrict__ ko, float* __restrict__ vo) {
  extern __shared__ char smem[];
  bf16* sA = (bf16*)smem;                       // 64 KB
  const int bx = blockIdx.x;                    // 0..23  (8 slabs x 3 mats)
  const int s = bx / 3, mat = bx % 3;
  const bf16*  pw  = (mat == 0) ? pq : ((mat == 1) ? pk : pv);
  const float* bb  = (mat == 0) ? bq : ((mat == 1) ? bk : bv);
  float*       out = (mat == 0) ? qo : ((mat == 1) ? ko : vo);
  const int tid = threadIdx.x, wave = tid >> 5, lane = tid & 31;
  const int n15 = lane & 15, hi8 = (lane & 16) >> 1;
  const size_t base = (size_t)s * NSEQ * DIM;

  for (int it = 0; it < 32; ++it) {
    int lin = it * 1024 + tid * 4;
    float4 f = *(const float4*)(x1 + base + lin);
    sA[lin + 0] = (bf16)f.x; sA[lin + 1] = (bf16)f.y;
    sA[lin + 2] = (bf16)f.z; sA[lin + 3] = (bf16)f.w;
  }
  __syncthreads();

  gemm_16tiles(sA, pw, 0, 8,
    [&](int nt) {
      float bias = bb[nt * 16 + n15];
      v8f acc;
#pragma unroll
      for (int r = 0; r < 8; ++r) acc[r] = bias;
      return acc;
    },
    [&](int nt, v8f acc) {
      int m0 = wave * 16 + hi8, n = nt * 16 + n15;
#pragma unroll
      for (int r = 0; r < 8; ++r) out[base + (size_t)(m0 + r) * DIM + n] = acc[r];
    });
}

// ------------------------------------------------------ fused edge slab kernel
// One workgroup = one (b,i) slab of 128x256. LDS: sA 64K bf16 | sE 128K f32 |
// sH 64K bf16 (256 KB of 320 KB/WGP).
__global__ __launch_bounds__(256)
void slab_kernel(const float* __restrict__ y,
                 const bf16* __restrict__ packWe,  const float* __restrict__ be,
                 const bf16* __restrict__ packWoe, const float* __restrict__ boe,
                 const bf16* __restrict__ packW1,  const float* __restrict__ b1,
                 const bf16* __restrict__ packW2,  const float* __restrict__ b2,
                 const float* __restrict__ qbuf, const float* __restrict__ kbuf,
                 const float* __restrict__ vbuf,
                 const float* __restrict__ ln4g, const float* __restrict__ ln4b,
                 const float* __restrict__ ln6g, const float* __restrict__ ln6b,
                 float* __restrict__ nodebuf, float* __restrict__ yout) {
  extern __shared__ char smem[];
  bf16*  sA = (bf16*)smem;                  // bf16 A operand (y / attn / y2)
  float* sE = (float*)(smem + 65536);       // f32 (e / attn_raw / edge / y2)
  bf16*  sH = (bf16*)(smem + 196608);       // bf16 hidden chunk / mlp out

  const int tid  = threadIdx.x;
  const int wave = tid >> 5;
  const int lane = tid & 31;
  const int slab = blockIdx.x;              // b*128 + i
  const int b    = slab >> 7;
  const size_t ybase = (size_t)slab * (NSEQ * DIM);
  const int rowbase  = (b << 7) * DIM;      // base row of k/v for this batch
  const int n15 = lane & 15;
  const int hi8 = (lane & 16) >> 1;

  // P0: stage y slab -> sA (bf16)
  for (int it = 0; it < 32; ++it) {
    int lin = it * 1024 + tid * 4;
    float4 f = *(const float4*)(y + ybase + lin);
    sA[lin + 0] = (bf16)f.x; sA[lin + 1] = (bf16)f.y;
    sA[lin + 2] = (bf16)f.z; sA[lin + 3] = (bf16)f.w;
  }
  __syncthreads();

  // P1: e = y @ We + be  -> sE (f32)
  gemm_16tiles(sA, packWe, 0, 8,
    [&](int nt) {
      float bias = be[nt * 16 + n15];
      v8f acc;
#pragma unroll
      for (int r = 0; r < 8; ++r) acc[r] = bias;
      return acc;
    },
    [&](int nt, v8f acc) {
      int m0 = wave * 16 + hi8, n = nt * 16 + n15;
#pragma unroll
      for (int r = 0; r < 8; ++r) sE[(m0 + r) * DIM + n] = acc[r];
    });
  __syncthreads();

  // P2: attn_raw = (q*k/sqrt(dk)) * (e+1) * e  -> sE (f32) + sA (bf16)
  {
    float qv = qbuf[(size_t)slab * DIM + tid] * 0.17677669529663688f; // 1/sqrt(32)
    for (int r = 0; r < NSEQ; ++r) {
      float kv = kbuf[rowbase + r * DIM + tid];
      int idx = r * DIM + tid;
      float e = sE[idx];
      float a = qv * kv * (e + 1.0f) * e;
      sE[idx] = a;
      sA[idx] = (bf16)a;
    }
  }
  __syncthreads();

  // P4: softmax over key axis (columns of slab) + node accumulation
  {
    float amax = -3.402823466e+38f;
    for (int r = 0; r < NSEQ; ++r) amax = fmaxf(amax, sE[r * DIM + tid]);
    float ssum = 0.f, nacc = 0.f;
    for (int r = 0; r < NSEQ; ++r) {
      float p = __expf(sE[r * DIM + tid] - amax);
      ssum += p;
      nacc += p * vbuf[rowbase + r * DIM + tid];
    }
    nodebuf[(size_t)slab * DIM + tid] = nacc / ssum;
  }
  __syncthreads();

  // P3: edge = attn_raw @ Woe + boe  -> sE (f32)
  gemm_16tiles(sA, packWoe, 0, 8,
    [&](int nt) {
      float bias = boe[nt * 16 + n15];
      v8f acc;
#pragma unroll
      for (int r = 0; r < 8; ++r) acc[r] = bias;
      return acc;
    },
    [&](int nt, v8f acc) {
      int m0 = wave * 16 + hi8, n = nt * 16 + n15;
#pragma unroll
      for (int r = 0; r < 8; ++r) sE[(m0 + r) * DIM + n] = acc[r];
    });
  __syncthreads();

  // P5: y2 = LN(edge + y)  -> sE (f32) + sA (bf16)   (wave-per-row LN)
  for (int jj = 0; jj < 16; ++jj) {
    int j = wave + jj * 8;
    float vreg[8], s = 0.f, s2 = 0.f;
#pragma unroll
    for (int cc = 0; cc < 8; ++cc) {
      int c = lane + cc * 32;
      float val = sE[j * DIM + c] + y[ybase + j * DIM + c];
      vreg[cc] = val; s += val; s2 += val * val;
    }
#pragma unroll
    for (int m = 16; m >= 1; m >>= 1) {
      s  += __shfl_xor(s, m, 32);
      s2 += __shfl_xor(s2, m, 32);
    }
    float mean = s * (1.f / DIM);
    float var  = s2 * (1.f / DIM) - mean * mean;
    float rs   = rsqrtf(var + 1e-5f);
#pragma unroll
    for (int cc = 0; cc < 8; ++cc) {
      int c = lane + cc * 32;
      float t = (vreg[cc] - mean) * rs * ln4g[c] + ln4b[c];
      sE[j * DIM + c] = t;
      sA[j * DIM + c] = (bf16)t;
    }
  }
  __syncthreads();

  // P6: MLP: out = relu(y2@W1 + b1) @ W2, hidden processed in 4 chunks of 256;
  //     output accumulators persist in VGPRs (16 tiles x v8f per wave).
  v8f accO[16];
#pragma unroll
  for (int nt = 0; nt < 16; ++nt)
#pragma unroll
    for (int r = 0; r < 8; ++r) accO[nt][r] = 0.f;

  for (int c4 = 0; c4 < 4; ++c4) {
    // GEMM1 chunk: h = relu(y2 @ W1[:, c4*256:(c4+1)*256] + b1) -> sH (bf16)
    gemm_16tiles(sA, packW1, c4 * 128, 8,
      [&](int nt) {
        float bias = b1[(c4 * 16 + nt) * 16 + n15];
        v8f acc;
#pragma unroll
        for (int r = 0; r < 8; ++r) acc[r] = bias;
        return acc;
      },
      [&](int nt, v8f acc) {
        int m0 = wave * 16 + hi8, n = nt * 16 + n15;
#pragma unroll
        for (int r = 0; r < 8; ++r) sH[(m0 + r) * DIM + n] = (bf16)fmaxf(acc[r], 0.f);
      });
    __syncthreads();
    // GEMM2 partial: accO += h @ W2[c4*256:(c4+1)*256, :]
    gemm_16tiles(sH, packW2, c4 * 8, 32,
      [&](int nt) { return accO[nt]; },
      [&](int nt, v8f acc) { accO[nt] = acc; });
    __syncthreads();
  }

  // scatter mlp-out -> sH (bf16)
  for (int nt = 0; nt < 16; ++nt) {
    int m0 = wave * 16 + hi8, n = nt * 16 + n15;
#pragma unroll
    for (int r = 0; r < 8; ++r) sH[(m0 + r) * DIM + n] = (bf16)accO[nt][r];
  }
  __syncthreads();

  // P7: y_out = LN(y2 + mlp_out + b2)  -> global
  for (int jj = 0; jj < 16; ++jj) {
    int j = wave + jj * 8;
    float vreg[8], s = 0.f, s2 = 0.f;
#pragma unroll
    for (int cc = 0; cc < 8; ++cc) {
      int c = lane + cc * 32;
      float val = sE[j * DIM + c] + (float)sH[j * DIM + c] + b2[c];
      vreg[cc] = val; s += val; s2 += val * val;
    }
#pragma unroll
    for (int m = 16; m >= 1; m >>= 1) {
      s  += __shfl_xor(s, m, 32);
      s2 += __shfl_xor(s2, m, 32);
    }
    float mean = s * (1.f / DIM);
    float var  = s2 * (1.f / DIM) - mean * mean;
    float rs   = rsqrtf(var + 1e-5f);
#pragma unroll
    for (int cc = 0; cc < 8; ++cc) {
      int c = lane + cc * 32;
      yout[ybase + j * DIM + c] = (vreg[cc] - mean) * rs * ln6g[c] + ln6b[c];
    }
  }
}

// ------------------------------------------------- node/x path (tiny, scalar)
__global__ __launch_bounds__(256)
void xpath_kernel(const float* __restrict__ x1, const float* __restrict__ node,
                  const float* __restrict__ Won, const float* __restrict__ bon,
                  const float* __restrict__ g3, const float* __restrict__ b3,
                  const float* __restrict__ W1, const float* __restrict__ b1,
                  const float* __restrict__ W2, const float* __restrict__ b2,
                  const float* __restrict__ g5, const float* __restrict__ b5,
                  float* __restrict__ xout) {
  __shared__ float sn[DIM];
  __shared__ float sx[DIM];
  __shared__ float sh[HID];
  __shared__ float red[DIM];
  const int row = blockIdx.x, t = threadIdx.x;

  sn[t] = node[(size_t)row * DIM + t];
  __syncthreads();
  float acc = bon[t];
  for (int k = 0; k < DIM; ++k) acc = fmaf(sn[k], Won[k * DIM + t], acc);
  float val = x1[(size_t)row * DIM + t] + acc;

  red[t] = val; __syncthreads();
  for (int off = 128; off > 0; off >>= 1) { if (t < off) red[t] += red[t + off]; __syncthreads(); }
  float mean = red[0] * (1.f / DIM); __syncthreads();
  float d = val - mean;
  red[t] = d * d; __syncthreads();
  for (int off = 128; off > 0; off >>= 1) { if (t < off) red[t] += red[t + off]; __syncthreads(); }
  float var = red[0] * (1.f / DIM); __syncthreads();
  float x2v = d * rsqrtf(var + 1e-5f) * g3[t] + b3[t];
  sx[t] = x2v; __syncthreads();

  for (int jj = 0; jj < 4; ++jj) {
    int j = t + 256 * jj;
    float h = b1[j];
    for (int k = 0; k < DIM; ++k) h = fmaf(sx[k], W1[k * HID + j], h);
    sh[j] = fmaxf(h, 0.f);
  }
  __syncthreads();
  float o = b2[t];
  for (int j = 0; j < HID; ++j) o = fmaf(sh[j], W2[j * DIM + t], o);
  float v2 = x2v + o;

  red[t] = v2; __syncthreads();
  for (int off = 128; off > 0; off >>= 1) { if (t < off) red[t] += red[t + off]; __syncthreads(); }
  float mean2 = red[0] * (1.f / DIM); __syncthreads();
  float d2 = v2 - mean2;
  red[t] = d2 * d2; __syncthreads();
  for (int off = 128; off > 0; off >>= 1) { if (t < off) red[t] += red[t + off]; __syncthreads(); }
  float var2 = red[0] * (1.f / DIM); __syncthreads();
  xout[(size_t)row * DIM + t] = d2 * rsqrtf(var2 + 1e-5f) * g5[t] + b5[t];
}

// --------------------------------------------------------------------- launch
extern "C" void kernel_launch(void* const* d_in, const int* in_sizes, int n_in,
                              void* d_out, int out_size, void* d_ws, size_t ws_size,
                              hipStream_t stream) {
  (void)in_sizes; (void)n_in; (void)out_size; (void)ws_size;
  const float* x    = (const float*)d_in[0];
  const float* y    = (const float*)d_in[1];
  const float* Wq   = (const float*)d_in[3];
  const float* bq   = (const float*)d_in[4];
  const float* Wk   = (const float*)d_in[5];
  const float* bk   = (const float*)d_in[6];
  const float* Wv   = (const float*)d_in[7];
  const float* bv   = (const float*)d_in[8];
  const float* We   = (const float*)d_in[9];
  const float* be   = (const float*)d_in[10];
  const float* Woe  = (const float*)d_in[11];
  const float* boe  = (const float*)d_in[12];
  const float* Won  = (const float*)d_in[13];
  const float* bon  = (const float*)d_in[14];
  const float* ln1g = (const float*)d_in[15];
  const float* ln1b = (const float*)d_in[16];
  const float* ln3g = (const float*)d_in[17];
  const float* ln3b = (const float*)d_in[18];
  const float* ln4g = (const float*)d_in[19];
  const float* ln4b = (const float*)d_in[20];
  const float* ln5g = (const float*)d_in[21];
  const float* ln5b = (const float*)d_in[22];
  const float* ln6g = (const float*)d_in[23];
  const float* ln6b = (const float*)d_in[24];
  const float* m1W1 = (const float*)d_in[25];
  const float* m1b1 = (const float*)d_in[26];
  const float* m1W2 = (const float*)d_in[27];
  const float* m1b2 = (const float*)d_in[28];
  const float* m2W1 = (const float*)d_in[29];
  const float* m2b1 = (const float*)d_in[30];
  const float* m2W2 = (const float*)d_in[31];
  const float* m2b2 = (const float*)d_in[32];

  char* ws = (char*)d_ws;
  bf16*  pWe  = (bf16*)(ws + 0);          // 128 KB
  bf16*  pWoe = (bf16*)(ws + 131072);     // 128 KB
  bf16*  pW1  = (bf16*)(ws + 262144);     // 512 KB (m2_W1)
  bf16*  pW2  = (bf16*)(ws + 786432);     // 512 KB (m2_W2)
  bf16*  pWq  = (bf16*)(ws + 1310720);    // 128 KB
  bf16*  pWk  = (bf16*)(ws + 1441792);    // 128 KB
  bf16*  pWv  = (bf16*)(ws + 1572864);    // 128 KB
  float* x1   = (float*)(ws + 1703936);   // 1 MB
  float* qb   = (float*)(ws + 2752512);   // 1 MB
  float* kb   = (float*)(ws + 3801088);   // 1 MB
  float* vb   = (float*)(ws + 4849664);   // 1 MB
  float* node = (float*)(ws + 5898240);   // 1 MB   (total ~6.6 MB)

  float* xout = (float*)d_out;
  float* yout = xout + (size_t)BATCH * NSEQ * DIM;

  // pre-pack weights into WMMA B-fragment layout (bf16)
  pack_kernel<<<256, 256, 0, stream>>>(We,   pWe,  256, 256);
  pack_kernel<<<256, 256, 0, stream>>>(Woe,  pWoe, 256, 256);
  pack_kernel<<<1024, 256, 0, stream>>>(m2W1, pW1, 256, 1024);
  pack_kernel<<<1024, 256, 0, stream>>>(m2W2, pW2, 1024, 256);
  pack_kernel<<<256, 256, 0, stream>>>(Wq,   pWq,  256, 256);
  pack_kernel<<<256, 256, 0, stream>>>(Wk,   pWk,  256, 256);
  pack_kernel<<<256, 256, 0, stream>>>(Wv,   pWv,  256, 256);

  lnx_kernel<<<128, 256, 0, stream>>>(x, ln1g, ln1b, x1);
  qkv_kernel<<<24, 256, 65536, stream>>>(x1, pWq, pWk, pWv, bq, bk, bv, qb, kb, vb);

  // fused edge pipeline: one WG per (b,i), 256 KB dynamic LDS
  slab_kernel<<<BATCH * NSEQ, 256, 262144, stream>>>(
      y, pWe, be, pWoe, boe, pW1, m2b1, pW2, m2b2,
      qb, kb, vb, ln4g, ln4b, ln6g, ln6b, node, yout);

  xpath_kernel<<<BATCH * NSEQ, 256, 0, stream>>>(
      x1, node, Won, bon, ln3g, ln3b, m1W1, m1b1, m1W2, m1b2, ln5g, ln5b, xout);
}
